// AgentNet_82308753260644
// MI455X (gfx1250) — compile-verified, matrix operand order checked by reading, not run on hardware
//
#include <hip/hip_runtime.h>

#define DEV_INLINE __device__ __forceinline__

typedef __attribute__((ext_vector_type(16))) __bf16 v16bf;
typedef __attribute__((ext_vector_type(8)))  float  v8f;
typedef __attribute__((ext_vector_type(4)))  unsigned int u32x4;
typedef __attribute__((ext_vector_type(8)))  int i32x8;
typedef __attribute__((ext_vector_type(4)))  int i32x4;

#if defined(__has_builtin)
#if __has_builtin(__builtin_amdgcn_tensor_load_to_lds) && __has_builtin(__builtin_amdgcn_s_wait_tensorcnt)
#define HAVE_TDM 1
#endif
#endif

constexpr int NN     = 50000;
constexpr int KK     = 16;
constexpr int DD     = 64;
constexpr int CC     = 40;
constexpr int NSTEPS = 4;
constexpr float DECAYF = 0.9f;

DEV_INLINE float4 ld4(const float* p) { return *reinterpret_cast<const float4*>(p); }

// ---------------- WMMA fragment helpers (CDNA5 16x16x32 bf16) ----------------
// A 16x32 (MxK) from row-major bf16 LDS tile:
// lanes 0-15: row M=lane, elems 0..7 = kb+0..7, 8..15 = kb+16..23
// lanes 16-31: row M=lane-16, elems 0..7 = kb+8..15, 8..15 = kb+24..31
DEV_INLINE v16bf load_a_frag(const __bf16* __restrict__ tile, int ld, int kb, int lane) {
  const int row = lane & 15;
  const int k0  = kb + ((lane & 16) ? 8 : 0);
  const __bf16* p = tile + row * ld + k0;
  v16bf a;
#pragma unroll
  for (int i = 0; i < 8; ++i) a[i] = p[i];
#pragma unroll
  for (int i = 0; i < 8; ++i) a[8 + i] = p[16 + i];
  return a;
}
// B fragments pre-swizzled in global memory: per (kb,nb) tile, per lane, 16
// contiguous bf16 (32 bytes) -> single wide load, no packing VALU.
DEV_INLINE v16bf load_b_frag_packed(const __bf16* __restrict__ w, int kTiles,
                                    int kb, int nb, int lane) {
  const size_t off = ((size_t)(nb >> 4) * kTiles + (kb >> 5)) * 512 + (size_t)lane * 16;
  return *reinterpret_cast<const v16bf*>(w + off);
}
DEV_INLINE v8f wmma_bf16(v16bf a, v16bf b, v8f c) {
  return __builtin_amdgcn_wmma_f32_16x16x32_bf16(false, a, false, b, (short)0, c, false, false);
}

// Pack f32 weights into WMMA B-fragment order (bf16), optional transpose/pad.
// normal:    B(k,n) = src[k*N + n]   (src is [K][N])
// transpose: B(k,n) = src[n*K + k]   (src is [N][K]) -- used for k_W^T
__global__ void pack_b_kernel(const float* __restrict__ src, __bf16* __restrict__ dst,
                              int K, int N, int Npad, int transpose)
{
  int i = blockIdx.x * blockDim.x + threadIdx.x;
  int tot = K * Npad;
  if (i >= tot) return;
  int e    = i & 15;
  int lane = (i >> 4) & 31;
  int f    = i >> 9;
  int kTiles = K >> 5;
  int kbi = f % kTiles;
  int nbi = f / kTiles;
  int k = kbi * 32 + ((lane & 16) ? 8 : 0) + ((e < 8) ? e : (8 + e));
  int n = nbi * 16 + (lane & 15);
  float v = 0.f;
  if (n < N) v = transpose ? src[(size_t)n * K + k] : src[(size_t)k * N + n];
  dst[i] = (__bf16)v;
}

// ---------------- fused 2-layer MLP: dst = resid + relu(LN?(in)@W1+b1)@W2+b2 --
template <int IN, int HID, int OUTP>
__global__ __launch_bounds__(128)
void mlp2_kernel(const float* __restrict__ src0,
                 const float* __restrict__ src1,
                 const int* __restrict__ gather_idx,
                 const __bf16* __restrict__ W1, const float* __restrict__ b1,
                 const __bf16* __restrict__ W2, const float* __restrict__ b2,
                 const float* __restrict__ resid,
                 float* __restrict__ dst, int dst_ld, int out_cols,
                 int n, int do_ln)
{
  __shared__ __bf16 lds_in [64 * IN];
  __shared__ __bf16 lds_hid[64 * HID];
  const int t = threadIdx.x;
  const int lane = t & 31;
  const int wv = t >> 5;
  const int r0 = blockIdx.x * 64;

  // stage 1: per-row (optional) layernorm -> bf16 LDS tile
  if (t < 64) {
    const int row = r0 + t;
    __bf16* ldr = lds_in + t * IN;
    if (row < n) {
      const float* p0;
      const float* p1 = nullptr;
      if (src1) {
        p0 = src0 + (size_t)row * 64;
        const int g = gather_idx ? gather_idx[row] : row;
        p1 = src1 + (size_t)g * 64;
      } else {
        p0 = src0 + (size_t)row * IN;
      }
      float mu = 0.f, inv = 1.f;
      if (do_ln) {
        float s = 0.f, s2 = 0.f;
        for (int j = 0; j < IN; j += 4) {
          const float* pp = (p1 && j >= 64) ? (p1 + (j - 64)) : (p0 + j);
          float4 v = ld4(pp);
          s  += v.x + v.y + v.z + v.w;
          s2 += v.x * v.x + v.y * v.y + v.z * v.z + v.w * v.w;
        }
        mu  = s * (1.f / IN);
        inv = rsqrtf(s2 * (1.f / IN) - mu * mu + 1e-5f);
      }
      for (int j = 0; j < IN; j += 4) {
        const float* pp = (p1 && j >= 64) ? (p1 + (j - 64)) : (p0 + j);
        float4 v = ld4(pp);
        ldr[j + 0] = (__bf16)((v.x - mu) * inv);
        ldr[j + 1] = (__bf16)((v.y - mu) * inv);
        ldr[j + 2] = (__bf16)((v.z - mu) * inv);
        ldr[j + 3] = (__bf16)((v.w - mu) * inv);
      }
    } else {
      for (int j = 0; j < IN; ++j) ldr[j] = (__bf16)0.f;
    }
  }
  __syncthreads();

  // stage 2: GEMM1 + bias + relu -> bf16 hidden tile
  {
    const __bf16* aT = lds_in + wv * 16 * IN;
    const int col  = lane & 15;
    const int roff = (lane & 16) ? 8 : 0;
    for (int nb = 0; nb < HID; nb += 16) {
      v8f acc = {};
#pragma unroll
      for (int kb = 0; kb < IN; kb += 32)
        acc = wmma_bf16(load_a_frag(aT, IN, kb, lane),
                        load_b_frag_packed(W1, IN / 32, kb, nb, lane), acc);
      const int c = nb + col;
      const float bias = b1[c];
#pragma unroll
      for (int j = 0; j < 8; ++j) {
        float v = fmaxf(acc[j] + bias, 0.f);
        lds_hid[(wv * 16 + roff + j) * HID + c] = (__bf16)v;
      }
    }
  }
  __syncthreads();

  // stage 3: GEMM2 + bias (+resid) -> global
  {
    const __bf16* aT = lds_hid + wv * 16 * HID;
    const int col  = lane & 15;
    const int roff = (lane & 16) ? 8 : 0;
    for (int nb = 0; nb < OUTP; nb += 16) {
      v8f acc = {};
#pragma unroll
      for (int kb = 0; kb < HID; kb += 32)
        acc = wmma_bf16(load_a_frag(aT, HID, kb, lane),
                        load_b_frag_packed(W2, HID / 32, kb, nb, lane), acc);
      const int c = nb + col;
      if (c < out_cols) {
        const float bias = b2[c];
#pragma unroll
        for (int j = 0; j < 8; ++j) {
          const int row = r0 + wv * 16 + roff + j;
          if (row < n) {
            float v = acc[j] + bias;
            if (resid) v += resid[(size_t)row * dst_ld + c];
            dst[(size_t)row * dst_ld + c] = v;
          }
        }
      }
    }
  }
}

// ---------------- q = ln(agent)@q_W+q_b ; w = q@k_W^T ; qdot = k_b . q -------
__global__ __launch_bounds__(128)
void qw_kernel(const float* __restrict__ agent, const __bf16* __restrict__ qWb,
               const float* __restrict__ q_b, const __bf16* __restrict__ kWTb,
               const float* __restrict__ k_b, float* __restrict__ wbuf,
               float* __restrict__ qdot, int n)
{
  __shared__ __bf16 la[64 * 64];
  __shared__ __bf16 lq[64 * 64];
  const int t = threadIdx.x, lane = t & 31, wv = t >> 5;
  const int r0 = blockIdx.x * 64;

  if (t < 64) {
    const int row = r0 + t;
    __bf16* ldr = la + t * 64;
    if (row < n) {
      const float* pr = agent + (size_t)row * 64;
      float s = 0.f, s2 = 0.f;
      for (int j = 0; j < 64; j += 4) {
        float4 v = ld4(pr + j);
        s += v.x + v.y + v.z + v.w;
        s2 += v.x * v.x + v.y * v.y + v.z * v.z + v.w * v.w;
      }
      float mu = s * (1.f / 64.f);
      float inv = rsqrtf(s2 * (1.f / 64.f) - mu * mu + 1e-5f);
      for (int j = 0; j < 64; j += 4) {
        float4 v = ld4(pr + j);
        ldr[j + 0] = (__bf16)((v.x - mu) * inv);
        ldr[j + 1] = (__bf16)((v.y - mu) * inv);
        ldr[j + 2] = (__bf16)((v.z - mu) * inv);
        ldr[j + 3] = (__bf16)((v.w - mu) * inv);
      }
    } else {
      for (int j = 0; j < 64; ++j) ldr[j] = (__bf16)0.f;
    }
  }
  __syncthreads();

  {  // q tile (no relu)
    const __bf16* aT = la + wv * 16 * 64;
    const int col = lane & 15, roff = (lane & 16) ? 8 : 0;
    for (int nb = 0; nb < 64; nb += 16) {
      v8f acc = {};
#pragma unroll
      for (int kb = 0; kb < 64; kb += 32)
        acc = wmma_bf16(load_a_frag(aT, 64, kb, lane),
                        load_b_frag_packed(qWb, 2, kb, nb, lane), acc);
      const int c = nb + col;
      const float bias = q_b[c];
#pragma unroll
      for (int j = 0; j < 8; ++j)
        lq[(wv * 16 + roff + j) * 64 + c] = (__bf16)(acc[j] + bias);
    }
  }
  __syncthreads();

  if (t < 64) {  // qdot = k_b . q
    const int row = r0 + t;
    if (row < n) {
      float s = 0.f;
      for (int c = 0; c < 64; ++c) s += (float)lq[t * 64 + c] * k_b[c];
      qdot[row] = s;
    }
  }

  {  // w = q @ k_W^T (pre-transposed fragment pack)
    const __bf16* aT = lq + wv * 16 * 64;
    const int col = lane & 15, roff = (lane & 16) ? 8 : 0;
    for (int nb = 0; nb < 128; nb += 16) {
      v8f acc = {};
#pragma unroll
      for (int kb = 0; kb < 64; kb += 32)
        acc = wmma_bf16(load_a_frag(aT, 64, kb, lane),
                        load_b_frag_packed(kWTb, 2, kb, nb, lane), acc);
      const int c = nb + col;
#pragma unroll
      for (int j = 0; j < 8; ++j) {
        const int row = r0 + wv * 16 + roff + j;
        if (row < n) wbuf[(size_t)row * 128 + c] = acc[j];
      }
    }
  }
}

// ---------------- msg = relu(ln(node)@msg_W + msg_b) -------------------------
// Input tile staged Global->LDS by the Tensor Data Mover when available.
__global__ __launch_bounds__(128)
void msg_kernel(const float* __restrict__ node, const __bf16* __restrict__ Wb,
                const float* __restrict__ bias, float* __restrict__ dst, int n)
{
  __shared__ float  stage[64 * 64];
  __shared__ __bf16 la[64 * 64];
  const int t = threadIdx.x, lane = t & 31, wv = t >> 5;
  const int r0 = blockIdx.x * 64;

#ifdef HAVE_TDM
  if (t < 32) {  // wave 0 issues one 2D TDM tile load: 64 rows x 64 f32
    const unsigned lds_off = (unsigned)(size_t)(void*)stage;  // low 32 bits = LDS offset
    const unsigned long long ga =
        (unsigned long long)(size_t)(node + (size_t)r0 * 64);
    int rows = n - r0; if (rows > 64) rows = 64;               // zero-fill past n
    u32x4 g0;
    g0[0] = 1u;                                                // count=1 user D#
    g0[1] = lds_off;                                           // lds_addr
    g0[2] = (unsigned)(ga & 0xFFFFFFFFu);                      // global_addr lo
    g0[3] = (unsigned)((ga >> 32) & 0x1FFFFFFu) | (2u << 30);  // addr hi | type=2
    i32x8 g1;
    g1[0] = (2 << 16);          // data_size=4B, mask=0, no flags
    g1[1] = (64 << 16);         // tensor_dim0[15:0]=64 (bits 79:48)
    g1[2] = (rows << 16);       // tensor_dim0[31:16]=0 | tensor_dim1[15:0]=rows
    g1[3] = (64 << 16);         // tensor_dim1[31:16]=0 | tile_dim0=64
    g1[4] = 64;                 // tile_dim1=64, tile_dim2=0
    g1[5] = 64;                 // tensor_dim0_stride=64 elements
    g1[6] = 0;
    g1[7] = 0;
    i32x4 g2 = {0, 0, 0, 0};
    i32x4 g3 = {0, 0, 0, 0};
#if defined(__clang_major__) && __clang_major__ >= 23
    i32x8 z8 = {0, 0, 0, 0, 0, 0, 0, 0};
    __builtin_amdgcn_tensor_load_to_lds(g0, g1, g2, g3, z8, 0);
#else
    __builtin_amdgcn_tensor_load_to_lds(g0, g1, g2, g3, 0);
#endif
    __builtin_amdgcn_s_wait_tensorcnt(0);
  }
  __syncthreads();
#else
  if (t < 64) {
    const int row = r0 + t;
    for (int j = 0; j < 64; j += 4) {
      float4 v = (row < n) ? ld4(node + (size_t)row * 64 + j) : float4{0.f, 0.f, 0.f, 0.f};
      *(float4*)(stage + t * 64 + j) = v;
    }
  }
  __syncthreads();
#endif

  if (t < 64) {  // layernorm from staged tile
    const int row = r0 + t;
    __bf16* ldr = la + t * 64;
    if (row < n) {
      const float* pr = stage + t * 64;
      float s = 0.f, s2 = 0.f;
      for (int j = 0; j < 64; ++j) { float v = pr[j]; s += v; s2 += v * v; }
      float mu = s * (1.f / 64.f);
      float inv = rsqrtf(s2 * (1.f / 64.f) - mu * mu + 1e-5f);
      for (int j = 0; j < 64; ++j) ldr[j] = (__bf16)((pr[j] - mu) * inv);
    } else {
      for (int j = 0; j < 64; ++j) ldr[j] = (__bf16)0.f;
    }
  }
  __syncthreads();

  const __bf16* aT = la + wv * 16 * 64;
  const int col = lane & 15, roff = (lane & 16) ? 8 : 0;
  for (int nb = 0; nb < 64; nb += 16) {
    v8f acc = {};
#pragma unroll
    for (int kb = 0; kb < 64; kb += 32)
      acc = wmma_bf16(load_a_frag(aT, 64, kb, lane),
                      load_b_frag_packed(Wb, 2, kb, nb, lane), acc);
    const int c = nb + col;
    const float b = bias[c];
#pragma unroll
    for (int j = 0; j < 8; ++j) {
      const int row = r0 + wv * 16 + roff + j;
      if (row < n) dst[(size_t)row * 64 + c] = fmaxf(acc[j] + b, 0.f);
    }
  }
}

// ---------------- attention scores with LN folded into the dot ---------------
__global__ void score_kernel(const float* __restrict__ node, const float* __restrict__ wbuf,
                             const float* __restrict__ qdot, const int* __restrict__ neighbors,
                             const int* __restrict__ pos, const int* __restrict__ prev,
                             const float* __restrict__ visited,
                             const float* __restrict__ attn_W, const float* __restrict__ attn_b,
                             const float* __restrict__ bias_params,
                             float* __restrict__ scores, int n)
{
  const int i = blockIdx.x * blockDim.x + threadIdx.x;
  if (i >= n * KK) return;
  const int a   = i >> 4;
  const int nbr = neighbors[i];
  const int cur = pos[a];
  const float* wa = wbuf + (size_t)a * 128;
  const float* hn = node + (size_t)nbr * 64;
  const float* hc = node + (size_t)cur * 64;
  float s = 0.f, s2 = 0.f, d = 0.f, sw = 0.f;
#pragma unroll 4
  for (int j = 0; j < 64; j += 4) {
    float4 h = ld4(hn + j); float4 w = ld4(wa + j);
    s  += h.x + h.y + h.z + h.w;
    s2 += h.x * h.x + h.y * h.y + h.z * h.z + h.w * h.w;
    d  += h.x * w.x + h.y * w.y + h.z * w.z + h.w * w.w;
    sw += w.x + w.y + w.z + w.w;
  }
#pragma unroll 4
  for (int j = 0; j < 64; j += 4) {
    float4 h = ld4(hc + j); float4 w = ld4(wa + 64 + j);
    s  += h.x + h.y + h.z + h.w;
    s2 += h.x * h.x + h.y * h.y + h.z * h.z + h.w * h.w;
    d  += h.x * w.x + h.y * w.y + h.z * w.z + h.w * w.w;
    sw += w.x + w.y + w.z + w.w;
  }
  const float mu  = s * (1.f / 128.f);
  const float var = s2 * (1.f / 128.f) - mu * mu;
  const float core = (d - mu * sw) * rsqrtf(var + 1e-5f);
  float sc = (core + qdot[a]) * 0.125f;              // scale = 1/sqrt(64)
  sc = sc * attn_W[0] + attn_b[0];
  if (nbr == cur)     sc += bias_params[1];          // stay_b
  if (nbr == prev[a]) sc += bias_params[0];          // back_b
  sc += (visited[nbr] > 0.5f) ? bias_params[2] : bias_params[3];
  scores[i] = sc;
}

__global__ void select_kernel(const float* __restrict__ scores, const int* __restrict__ neighbors,
                              int* __restrict__ npos, int n)
{
  const int a = blockIdx.x * blockDim.x + threadIdx.x;
  if (a >= n) return;
  const float* sc = scores + (size_t)a * KK;
  float best = sc[0]; int bi = 0;
#pragma unroll
  for (int k = 1; k < KK; ++k) { float v = sc[k]; if (v > best) { best = v; bi = k; } }
  npos[a] = neighbors[(size_t)a * KK + bi];
}

// ---------------- scatter / gather / elementwise -----------------------------
__global__ void zero_kernel(float* __restrict__ p, int tot) {
  int i = blockIdx.x * blockDim.x + threadIdx.x;
  if (i < tot) p[i] = 0.f;
}
__global__ void segment_kernel(const float* __restrict__ agent, const int* __restrict__ npos,
                               float* __restrict__ agg, int n)
{
  int i = blockIdx.x * blockDim.x + threadIdx.x;
  if (i >= n * DD) return;
  int a = i >> 6, d = i & 63;
  atomicAdd(&agg[(size_t)npos[a] * DD + d], agent[i]);
}
__global__ void conv_agg_kernel(const float* __restrict__ msg, const int* __restrict__ neighbors,
                                float* __restrict__ out, int n)
{
  int i = blockIdx.x * blockDim.x + threadIdx.x;
  if (i >= n * DD) return;
  int nd = i >> 6, d = i & 63;
  const int* nb = neighbors + (size_t)nd * KK;
  float s = 0.f;
#pragma unroll 4
  for (int k = 0; k < KK; ++k) {
    if (k + 1 < KK) __builtin_prefetch(msg + (size_t)nb[k + 1] * DD + d, 0, 0);
    s += msg[(size_t)nb[k] * DD + d];
  }
  out[i] = s * (1.f / 16.f);
}
__global__ void decay_kernel(float* __restrict__ visited, int n) {
  int i = blockIdx.x * blockDim.x + threadIdx.x;
  if (i < n) visited[i] *= DECAYF;
}
__global__ void setv_kernel(float* __restrict__ visited, const int* __restrict__ npos, int n) {
  int i = blockIdx.x * blockDim.x + threadIdx.x;
  if (i < n) visited[npos[i]] = 1.f;
}
__global__ void init_kernel(const float* __restrict__ agent_table, float* __restrict__ agent,
                            int* __restrict__ pos, int* __restrict__ prev,
                            float* __restrict__ visited, int n)
{
  int i = blockIdx.x * blockDim.x + threadIdx.x;
  if (i < n * DD) agent[i] = agent_table[i];
  if (i < n) { pos[i] = i; prev[i] = i; visited[i] = 1.f; }
}

// ---------------- host orchestration ----------------------------------------
extern "C" void kernel_launch(void* const* d_in, const int* in_sizes, int n_in,
                              void* d_out, int out_size, void* d_ws, size_t ws_size,
                              hipStream_t stream)
{
  (void)in_sizes; (void)n_in; (void)out_size; (void)ws_size;
  const float* x           = (const float*)d_in[0];
  const float* agent_table = (const float*)d_in[1];
  const float* in_W1 = (const float*)d_in[2];
  const float* in_b1 = (const float*)d_in[3];
  const float* in_W2 = (const float*)d_in[4];
  const float* in_b2 = (const float*)d_in[5];
  const float* q_W   = (const float*)d_in[6];
  const float* q_b   = (const float*)d_in[7];
  const float* k_W   = (const float*)d_in[8];
  const float* k_b   = (const float*)d_in[9];
  const float* attn_W      = (const float*)d_in[10];
  const float* attn_b      = (const float*)d_in[11];
  const float* bias_params = (const float*)d_in[12];
  const float* msg_W = (const float*)d_in[13];
  const float* msg_b = (const float*)d_in[14];
  const float* conv_W1 = (const float*)d_in[15];
  const float* conv_b1 = (const float*)d_in[16];
  const float* conv_W2 = (const float*)d_in[17];
  const float* conv_b2 = (const float*)d_in[18];
  const float* node_W1 = (const float*)d_in[19];
  const float* node_b1 = (const float*)d_in[20];
  const float* node_W2 = (const float*)d_in[21];
  const float* node_b2 = (const float*)d_in[22];
  const float* agent_W1 = (const float*)d_in[23];
  const float* agent_b1 = (const float*)d_in[24];
  const float* agent_W2 = (const float*)d_in[25];
  const float* agent_b2 = (const float*)d_in[26];
  const float* ro_W1 = (const float*)d_in[27];
  const float* ro_b1 = (const float*)d_in[28];
  const float* ro_W2 = (const float*)d_in[29];
  const float* ro_b2 = (const float*)d_in[30];
  const int* neighbors = (const int*)d_in[31];

  char* p = (char*)d_ws;
  auto carve = [&](size_t bytes) -> char* {
    char* q = p; p += (bytes + 255) & ~(size_t)255; return q;
  };
  float* node   = (float*)carve((size_t)NN * DD * 4);
  float* agent  = (float*)carve((size_t)NN * DD * 4);
  float* agg    = (float*)carve((size_t)NN * DD * 4);  // reused as conv_agg
  float* msg    = (float*)carve((size_t)NN * DD * 4);
  float* wbuf   = (float*)carve((size_t)NN * 128 * 4);
  float* qdot   = (float*)carve((size_t)NN * 4);
  float* scores = (float*)carve((size_t)NN * KK * 4);
  float* visited= (float*)carve((size_t)NN * 4);
  int* pb0 = (int*)carve((size_t)NN * 4);
  int* pb1 = (int*)carve((size_t)NN * 4);
  int* pb2 = (int*)carve((size_t)NN * 4);
  __bf16* inW1b   = (__bf16*)carve(128 * 128 * 2);
  __bf16* inW2b   = (__bf16*)carve(128 * 64 * 2);
  __bf16* qWb     = (__bf16*)carve(64 * 64 * 2);
  __bf16* kWTb    = (__bf16*)carve(64 * 128 * 2);    // k_W^T packed
  __bf16* msgWb   = (__bf16*)carve(64 * 64 * 2);
  __bf16* convW1b = (__bf16*)carve(128 * 256 * 2);
  __bf16* convW2b = (__bf16*)carve(256 * 64 * 2);
  __bf16* nodeW1b = (__bf16*)carve(128 * 256 * 2);
  __bf16* nodeW2b = (__bf16*)carve(256 * 64 * 2);
  __bf16* agW1b   = (__bf16*)carve(128 * 256 * 2);
  __bf16* agW2b   = (__bf16*)carve(256 * 64 * 2);
  __bf16* roW1b   = (__bf16*)carve(64 * 128 * 2);
  __bf16* roW2b   = (__bf16*)carve(128 * 48 * 2);    // 40 padded to 48

  auto pk = [&](const float* s, __bf16* d, int Kd, int Nd, int Npad, int tr) {
    int tot = Kd * Npad;
    pack_b_kernel<<<(tot + 255) / 256, 256, 0, stream>>>(s, d, Kd, Nd, Npad, tr);
  };
  pk(in_W1, inW1b, 128, 128, 128, 0);
  pk(in_W2, inW2b, 128, 64, 64, 0);
  pk(q_W,   qWb,   64, 64, 64, 0);
  pk(k_W,   kWTb,  64, 128, 128, 1);   // B(k,n) = k_W[n,k]
  pk(msg_W, msgWb, 64, 64, 64, 0);
  pk(conv_W1, convW1b, 128, 256, 256, 0);
  pk(conv_W2, convW2b, 256, 64, 64, 0);
  pk(node_W1, nodeW1b, 128, 256, 256, 0);
  pk(node_W2, nodeW2b, 256, 64, 64, 0);
  pk(agent_W1, agW1b, 128, 256, 256, 0);
  pk(agent_W2, agW2b, 256, 64, 64, 0);
  pk(ro_W1, roW1b, 64, 128, 128, 0);
  pk(ro_W2, roW2b, 128, 40, 48, 0);

  const int gE  = (NN * DD + 255) / 256;   // elementwise over N*D
  const int gN  = (NN + 255) / 256;        // elementwise over N
  const int gNK = (NN * KK + 255) / 256;   // elementwise over N*K
  const int gT  = (NN + 63) / 64;          // 64-row GEMM tiles

  init_kernel<<<gE, 256, 0, stream>>>(agent_table, agent, pb0, pb1, visited, NN);

  // node = relu(x@in_W1+b1)@in_W2+b2
  mlp2_kernel<128, 128, 64><<<gT, 128, 0, stream>>>(
      x, nullptr, nullptr, inW1b, in_b1, inW2b, in_b2,
      nullptr, node, 64, 64, NN, 0);

  int* pos = pb0; int* prv = pb1; int* npos = pb2;
  for (int s = 0; s < NSTEPS; ++s) {
    qw_kernel<<<gT, 128, 0, stream>>>(agent, qWb, q_b, kWTb, k_b, wbuf, qdot, NN);
    score_kernel<<<gNK, 256, 0, stream>>>(node, wbuf, qdot, neighbors, pos, prv, visited,
                                          attn_W, attn_b, bias_params, scores, NN);
    select_kernel<<<gN, 256, 0, stream>>>(scores, neighbors, npos, NN);
    // agent += MLP(ln(concat(agent, node[new_pos])))
    mlp2_kernel<128, 256, 64><<<gT, 128, 0, stream>>>(
        agent, node, npos, agW1b, agent_b1, agW2b, agent_b2,
        agent, agent, 64, 64, NN, 1);
    // agg = segment_sum(agent, new_pos)
    zero_kernel<<<gE, 256, 0, stream>>>(agg, NN * DD);
    segment_kernel<<<gE, 256, 0, stream>>>(agent, npos, agg, NN);
    // node += MLP(ln(concat(node, agg)))
    mlp2_kernel<128, 256, 64><<<gT, 128, 0, stream>>>(
        node, agg, nullptr, nodeW1b, node_b1, nodeW2b, node_b2,
        node, node, 64, 64, NN, 1);
    // msg + sparse conv
    msg_kernel<<<gT, 128, 0, stream>>>(node, msgWb, msg_b, msg, NN);
    conv_agg_kernel<<<gE, 256, 0, stream>>>(msg, neighbors, agg, NN);
    mlp2_kernel<128, 256, 64><<<gT, 128, 0, stream>>>(
        node, agg, nullptr, convW1b, conv_b1, convW2b, conv_b2,
        node, node, 64, 64, NN, 1);
    // visited = (visited*DECAY).at[new_pos].set(1)
    decay_kernel<<<gN, 256, 0, stream>>>(visited, NN);
    setv_kernel<<<gN, 256, 0, stream>>>(visited, npos, NN);
    int* t2 = prv; prv = pos; pos = npos; npos = t2;
  }

  // out = relu(node@ro_W1+b1)@ro_W2+b2  (40 cols, W2 padded to 48)
  mlp2_kernel<64, 128, 48><<<gT, 128, 0, stream>>>(
      node, nullptr, nullptr, roW1b, ro_b1, roW2b, ro_b2,
      nullptr, (float*)d_out, CC, CC, NN, 0);
}
